// SynthesisNetwork_25262997635843
// MI455X (gfx1250) — matrix operand order repaired and checked by reading, not compiled, not run on hardware
//
#include <hip/hip_runtime.h>
#include <hip/hip_bf16.h>
#include <cmath>
#include <cstdint>

// ---------------------------------------------------------------------------
// CDNA5 WMMA types (wave32): 16x16x32 bf16 -> f32
// ---------------------------------------------------------------------------
typedef float  v8f   __attribute__((ext_vector_type(8)));
typedef __bf16 v16bf __attribute__((ext_vector_type(16)));

union Frag { uint4 q[2]; v16bf v; };

// ---- optional CDNA5 async global->LDS path (guarded; falls back cleanly) ----
#if defined(__has_builtin)
#  if __has_builtin(__builtin_amdgcn_global_load_async_to_lds_b128)
#    define HAVE_ASYNC 1
#  else
#    define HAVE_ASYNC 0
#  endif
#else
#  define HAVE_ASYNC 0
#endif

template <int N>
__device__ __forceinline__ void wait_asynccnt() {
#if HAVE_ASYNC
#  if defined(__has_builtin) && __has_builtin(__builtin_amdgcn_s_wait_asynccnt)
  __builtin_amdgcn_s_wait_asynccnt(N);
#  else
  asm volatile("s_wait_asynccnt %0" ::"i"(N) : "memory");
#  endif
#endif
}

#if HAVE_ASYNC
// Builtin signature (from compiler diagnostic): arg0 is AS1 pointer to a
// 16-byte int vector (non-const); arg1 the LDS-side equivalent (AS3).
typedef int v4i_t __attribute__((vector_size(16)));
typedef __attribute__((address_space(1))) v4i_t* as1_v4i;
typedef __attribute__((address_space(3))) v4i_t* as3_v4i;

// Stage one 32K x 64N bf16 B-tile (4KB: two contiguous 2KB slabs in the staged
// weight buffer) into this wave's private LDS buffer. 8 async b128 issues/wave.
__device__ __forceinline__ void stageB_async(const unsigned short* gbase,
                                             unsigned short* lbase,
                                             int lane, int slabStrideBytes) {
  const char* g = (const char*)gbase;
  char* l = (char*)lbase;
  const int byteOff = lane * 16;
#pragma unroll
  for (int i = 0; i < 8; ++i) {
    const int tb = i * 512 + byteOff;   // byte within 4KB tile
    const int slab = tb >> 11;          // which kblk row (0/1)
    const int rel = tb & 2047;
    void* gsrc = (void*)(g + (size_t)slab * slabStrideBytes + rel);
    void* ldst = (void*)(l + tb);
    __builtin_amdgcn_global_load_async_to_lds_b128((as1_v4i)gsrc, (as3_v4i)ldst, 0, 0);
  }
}
#endif

__device__ __forceinline__ unsigned short f2bf(float f) {
  unsigned int u = __float_as_uint(f);
  u += 0x7FFFu + ((u >> 16) & 1u);              // round-to-nearest-even
  return (unsigned short)(u >> 16);
}
__device__ __forceinline__ int imin(int a, int b) { return a < b ? a : b; }
__device__ __forceinline__ int imax(int a, int b) { return a > b ? a : b; }

// jax.image.resize(..., 'bilinear') 2x upsample: half-pixel centers,
// out o maps to src coord o/2 - 0.25 -> weights {0.25, 0.75}, edge clamped.
__device__ __forceinline__ float bilerp2x(const float* __restrict__ src,
                                          int b, int h, int w, int c,
                                          int Hs, int Ws, int C) {
  int ty = h >> 1, tx = w >> 1;
  int y0, y1, x0, x1; float wy0, wy1, wx0, wx1;
  if (h & 1) { y0 = ty; y1 = imin(ty + 1, Hs - 1); wy0 = 0.75f; wy1 = 0.25f; }
  else       { y0 = imax(ty - 1, 0); y1 = ty;      wy0 = 0.25f; wy1 = 0.75f; }
  if (w & 1) { x0 = tx; x1 = imin(tx + 1, Ws - 1); wx0 = 0.75f; wx1 = 0.25f; }
  else       { x0 = imax(tx - 1, 0); x1 = tx;      wx0 = 0.25f; wx1 = 0.75f; }
  const float* base = src + (size_t)b * Hs * Ws * C;
  float v00 = base[(y0 * Ws + x0) * C + c];
  float v01 = base[(y0 * Ws + x1) * C + c];
  float v10 = base[(y1 * Ws + x0) * C + c];
  float v11 = base[(y1 * Ws + x1) * C + c];
  return wy0 * (wx0 * v00 + wx1 * v01) + wy1 * (wx0 * v10 + wx1 * v11);
}

// ---------------------------------------------------------------------------
// Styles: s[b,i] = w[b,:] @ style_w[:,i] + style_b[i]   (B=32, K=512)
// ---------------------------------------------------------------------------
__global__ void style_kernel(const float* __restrict__ w,
                             const float* __restrict__ sw,
                             const float* __restrict__ sb,
                             float* __restrict__ s, int cin) {
  int idx = blockIdx.x * blockDim.x + threadIdx.x;
  if (idx >= 32 * cin) return;
  int b = idx / cin, i = idx - b * cin;
  const float* wr = w + b * 512;
  float acc = sb[i];
  for (int l = 0; l < 512; ++l) acc += wr[l] * sw[l * cin + i];
  s[idx] = acc;
}

// wsq[i,n] = wscale^2 * sum_{3x3} conv_w[ky,kx,i,n]^2
__global__ void wsumsq_kernel(const float* __restrict__ cw,
                              float* __restrict__ out,
                              int cin, int cout, float ws2) {
  int idx = blockIdx.x * blockDim.x + threadIdx.x;
  if (idx >= cin * cout) return;
  int i = idx / cout, n = idx - i * cout;
  float a = 0.f;
  for (int kk = 0; kk < 9; ++kk) {
    float v = cw[(kk * cin + i) * cout + n];
    a += v * v;
  }
  out[idx] = a * ws2;
}

// rsig[b,n] = 1/sqrt( sum_i s[b,i]^2 * wsq[i,n] + 1e-8 )
__global__ void rsigma_kernel(const float* __restrict__ s,
                              const float* __restrict__ wsq,
                              float* __restrict__ rsig,
                              int cin, int cout) {
  int idx = blockIdx.x * blockDim.x + threadIdx.x;
  if (idx >= 32 * cout) return;
  int b = idx / cout, n = idx - b * cout;
  const float* sr = s + b * cin;
  float a = 1e-8f;
  for (int i = 0; i < cin; ++i) {
    float sv = sr[i];
    a += sv * sv * wsq[i * cout + n];
  }
  rsig[idx] = rsqrtf(a);
}

// Stage weights into WMMA B-operand order, bf16:
//   wst[(kblk*Cout + n)*16 + j] = wscale * conv_w[ky,kx,i,n],  K = kblk*16+j,
//   K = (ky*3+kx)*Cin + i.  (16 consecutive K for a fixed column n, contiguous)
__global__ void wstage_kernel(const float* __restrict__ cw,
                              unsigned short* __restrict__ wst,
                              int cin, int cout, float wscale) {
  int idx = blockIdx.x * blockDim.x + threadIdx.x;
  if (idx >= 9 * cin * cout) return;
  int j = idx & 15;
  int t = idx >> 4;
  int n = t % cout;
  int kblk = t / cout;
  int K = kblk * 16 + j;
  int kk = K / cin;
  int i = K - kk * cin;
  wst[idx] = f2bf(cw[(kk * cin + i) * cout + n] * wscale);
}

// Stage activations: xq[b,h,w,i] = bf16( X(b,h,w,i) * s[b,i] )
// mode 0: src is const param [4,4,C] (broadcast over batch)
// mode 1: src is [B,H,W,C] fp32
// mode 2: src is [B,H/2,W/2,C] fp32, bilinear 2x upsample fused here
__global__ void xstage_kernel(const float* __restrict__ src,
                              const float* __restrict__ s,
                              unsigned short* __restrict__ xq,
                              int H, int W, int C, int mode) {
  int idx = blockIdx.x * blockDim.x + threadIdx.x;
  if (idx >= 32 * H * W * C) return;
  int c = idx % C;
  int t = idx / C;
  int w = t % W; t /= W;
  int h = t % H;
  int b = t / H;
  float v;
  if (mode == 0)      v = src[(h * W + w) * C + c];
  else if (mode == 1) v = src[idx];
  else                v = bilerp2x(src, b, h, w, c, H >> 1, W >> 1, C);
  xq[idx] = f2bf(v * s[b * C + c]);
}

// ---------------------------------------------------------------------------
// Implicit-GEMM modulated 3x3 conv ('SAME').
//   One wave owns a 16M x 64N tile: 4 accumulators share one A fragment
//   (4 WMMAs per 32-K step). B tiles (4KB/step) are double-buffered in LDS
//   via async global->LDS loads when available (private per wave, no barriers;
//   s_wait_asynccnt 8 guarantees the in-flight previous step completed since
//   async loads retire in order). Fallback: direct global B reads (identical
//   addressing; the LDS tile mirrors the staged-global layout).
// Epilogue: y = acc * rsig[b,n] + nstr[n]*noise[b,h,w] + bias[n]; leaky 0.2.
// ---------------------------------------------------------------------------
__global__ __launch_bounds__(128)
void conv_wmma_kernel(const unsigned short* __restrict__ xq,
                      const unsigned short* __restrict__ wst,
                      const float* __restrict__ rsig,
                      const float* __restrict__ nstr,
                      const float* __restrict__ bias,
                      const float* __restrict__ noise,
                      float* __restrict__ xout,
                      int H, int W, int Cin, int Cout) {
#if HAVE_ASYNC
  __shared__ unsigned short ldsB[4 * 2 * 2048];   // 4 waves x 2 bufs x 4KB = 32KB
#endif
  const int lane = threadIdx.x & 31;
  const int wv   = threadIdx.x >> 5;
  const int M    = H * W;
  const int nt64 = Cout >> 6;                     // every layer: Cout % 64 == 0
  const int tilesPerImg = (M >> 4) * nt64;
  const int tile = blockIdx.x * 4 + wv;           // wave-uniform
  if (tile >= 32 * tilesPerImg) return;           // wave-uniform: EXEC stays full
  const int b  = tile / tilesPerImg;
  const int tr = tile - b * tilesPerImg;
  const int mt = tr / nt64;
  const int m0 = mt << 4;
  const int n0 = (tr - mt * nt64) << 6;

  const int half = lane >> 4;                     // K-half per WMMA reg layout
  const int r    = lane & 15;
  const int m    = m0 + r;
  const int hM   = m / W, wMc = m - hM * W;

  const int nsteps = (9 * Cin) >> 5;
  const int slabStride = Cout * 32;               // bytes between kblk rows
#if HAVE_ASYNC
  unsigned short* lb0 = ldsB + wv * 4096;         // this wave's 2 x 2048-ushort bufs
  stageB_async(wst + ((size_t)0 * Cout + n0) * 16, lb0, lane, slabStride);
#endif

  v8f acc0 = {0.f, 0.f, 0.f, 0.f, 0.f, 0.f, 0.f, 0.f};
  v8f acc1 = acc0, acc2 = acc0, acc3 = acc0;

  int t = 0;
  for (int kk = 0; kk < 9; ++kk) {
    const int dy = kk / 3 - 1;
    const int dx = kk - (kk / 3) * 3 - 1;
    const int hh = hM + dy, ww = wMc + dx;
    const bool inb = ((unsigned)hh < (unsigned)H) && ((unsigned)ww < (unsigned)W);
    const unsigned short* arow =
        xq + (size_t)((b * H + hh) * W + ww) * Cin + half * 16;  // valid iff inb
    for (int ib = 0; ib < Cin; ib += 32, ++t) {
#if HAVE_ASYNC
      if (t + 1 < nsteps) {
        stageB_async(wst + ((size_t)(2 * (t + 1)) * Cout + n0) * 16,
                     lb0 + ((t + 1) & 1) * 2048, lane, slabStride);
        wait_asynccnt<8>();                       // step t complete (in-order)
      } else {
        wait_asynccnt<0>();
      }
#endif
      // A fragment: lane holds K rel {0..15} (half 0) / {16..31}
      // (half 1); NHWC channel-contiguous -> two 16B loads.
      Frag fa;
      if (inb) {
        const uint4* pa = (const uint4*)(arow + ib);
        fa.q[0] = pa[0];
        fa.q[1] = pa[1];                          // +8 bf16
      } else {
        fa.q[0] = make_uint4(0u, 0u, 0u, 0u);
        fa.q[1] = make_uint4(0u, 0u, 0u, 0u);
      }
      // B fragments: lane = column (n0 + s*16 + r), 16 consecutive K each.
      // Subtile stride = 256 ushorts in both LDS tile and staged global.
#if HAVE_ASYNC
      const unsigned short* bb = lb0 + (t & 1) * 2048 + half * 1024 + r * 16;
#else
      const int k0 = kk * Cin + ib;
      const unsigned short* bb =
          wst + ((size_t)((k0 >> 4) + half) * Cout + n0 + r) * 16;
#endif
      Frag f0, f1, f2, f3;
      { const uint4* p = (const uint4*)(bb +   0); f0.q[0] = p[0]; f0.q[1] = p[1]; }
      { const uint4* p = (const uint4*)(bb + 256); f1.q[0] = p[0]; f1.q[1] = p[1]; }
      { const uint4* p = (const uint4*)(bb + 512); f2.q[0] = p[0]; f2.q[1] = p[1]; }
      { const uint4* p = (const uint4*)(bb + 768); f3.q[0] = p[0]; f3.q[1] = p[1]; }
      acc0 = __builtin_amdgcn_wmma_f32_16x16x32_bf16(false, fa.v, false, f0.v, (short)0, acc0, false, false);
      acc1 = __builtin_amdgcn_wmma_f32_16x16x32_bf16(false, fa.v, false, f1.v, (short)0, acc1, false, false);
      acc2 = __builtin_amdgcn_wmma_f32_16x16x32_bf16(false, fa.v, false, f2.v, (short)0, acc2, false, false);
      acc3 = __builtin_amdgcn_wmma_f32_16x16x32_bf16(false, fa.v, false, f3.v, (short)0, acc3, false, false);
    }
  }

  // D layout: lane (lo|hi) -> col (lane&15); vgpr rr -> row m0+rr(+8 hi half)
#pragma unroll
  for (int s = 0; s < 4; ++s) {
    v8f a = (s == 0) ? acc0 : (s == 1) ? acc1 : (s == 2) ? acc2 : acc3;
    const int n = n0 + s * 16 + r;
    const float rs  = rsig[b * Cout + n];
    const float nsv = nstr[n];
    const float bv  = bias[n];
#pragma unroll
    for (int rr = 0; rr < 8; ++rr) {
      const int mm = m0 + rr + half * 8;
      const int h2 = mm / W, w2 = mm - h2 * W;
      float v = a[rr] * rs + nsv * noise[(b * H + h2) * W + w2] + bv;
      v = v > 0.f ? v : 0.2f * v;
      xout[(size_t)((b * H + h2) * W + w2) * Cout + n] = v;
    }
  }
}

// to_rgb (1x1 modulated conv, no demod) + optional bilinear-upsampled prev rgb
__global__ void torgb_kernel(const float* __restrict__ x,
                             const float* __restrict__ s,
                             const float* __restrict__ cw,
                             const float* __restrict__ bias,
                             const float* __restrict__ prev,
                             float* __restrict__ out,
                             int H, int W, int cin, float wscale, int hasPrev) {
  int idx = blockIdx.x * blockDim.x + threadIdx.x;
  if (idx >= 32 * H * W * 3) return;
  int c = idx % 3;
  int t = idx / 3;
  int w = t % W; t /= W;
  int h = t % H;
  int b = t / H;
  const float* xr = x + (size_t)((b * H + h) * W + w) * cin;
  const float* sr = s + b * cin;
  float dot = 0.f;
  for (int i = 0; i < cin; ++i) dot += xr[i] * sr[i] * cw[i * 3 + c];
  float v = dot * wscale + bias[c];
  if (hasPrev) v += bilerp2x(prev, b, h, w, c, H >> 1, W >> 1, 3);
  out[idx] = v;
}

// ---------------------------------------------------------------------------
// Host orchestration
// ---------------------------------------------------------------------------
extern "C" void kernel_launch(void* const* d_in, const int* in_sizes, int n_in,
                              void* d_out, int out_size, void* d_ws, size_t ws_size,
                              hipStream_t stream) {
  (void)out_size; (void)ws_size;
  static const int CIN[9]    = {512, 512, 256, 256, 128, 128, 64, 64, 64};
  static const int COUT[9]   = {512, 256, 256, 128, 128, 64, 64, 64, 64};
  static const int HH[9]     = {4, 8, 8, 16, 16, 32, 32, 64, 64};
  static const int UP[9]     = {0, 1, 0, 1, 0, 1, 0, 1, 0};
  static const int RGBCIN[5] = {512, 256, 128, 64, 64};

  const float* lat = (const float*)d_in[0];
  const float* noiseP[9];
  for (int j = 0; j < 9; ++j) noiseP[j] = (const float*)d_in[1 + j];

  // params may be flattened pytree-sorted (bias,conv_w,...; const after blocks)
  // or insertion-ordered (const first; conv_w,style_w,...). in_sizes[10]
  // disambiguates: insertion order puts const (4*4*512) there.
  const bool ins = (n_in > 10) && (in_sizes[10] == 4 * 4 * 512);
  const float *bConv[9], *bStyW[9], *bStyB[9], *bNstr[9], *bBias[9];
  const float *rConv[5], *rStyW[5], *rStyB[5], *rBias[5];
  const float* cst;
  if (ins) {
    cst = (const float*)d_in[10];
    for (int j = 0; j < 9; ++j) {
      int base = 11 + 5 * j;
      bConv[j] = (const float*)d_in[base + 0];
      bStyW[j] = (const float*)d_in[base + 1];
      bStyB[j] = (const float*)d_in[base + 2];
      bNstr[j] = (const float*)d_in[base + 3];
      bBias[j] = (const float*)d_in[base + 4];
    }
    for (int r2 = 0; r2 < 5; ++r2) {
      int base = 56 + 4 * r2;
      rConv[r2] = (const float*)d_in[base + 0];
      rStyW[r2] = (const float*)d_in[base + 1];
      rStyB[r2] = (const float*)d_in[base + 2];
      rBias[r2] = (const float*)d_in[base + 3];
    }
  } else {
    for (int j = 0; j < 9; ++j) {
      int base = 10 + 5 * j;
      bBias[j] = (const float*)d_in[base + 0];
      bConv[j] = (const float*)d_in[base + 1];
      bNstr[j] = (const float*)d_in[base + 2];
      bStyB[j] = (const float*)d_in[base + 3];
      bStyW[j] = (const float*)d_in[base + 4];
    }
    cst = (const float*)d_in[55];
    for (int r2 = 0; r2 < 5; ++r2) {
      int base = 56 + 4 * r2;
      rBias[r2] = (const float*)d_in[base + 0];
      rConv[r2] = (const float*)d_in[base + 1];
      rStyB[r2] = (const float*)d_in[base + 2];
      rStyW[r2] = (const float*)d_in[base + 3];
    }
  }

  // ---- workspace bump allocator (256B aligned) ----
  uintptr_t cur = (uintptr_t)d_ws;
  cur = (cur + 255) & ~(uintptr_t)255;
  auto alloc = [&](size_t bytes) -> void* {
    void* p = (void*)cur;
    cur += (bytes + 255) & ~(size_t)255;
    return p;
  };
  float* sBlk[9];
  for (int j = 0; j < 9; ++j) sBlk[j] = (float*)alloc((size_t)32 * CIN[j] * 4);
  float* sRgb[5];
  for (int r2 = 0; r2 < 5; ++r2) sRgb[r2] = (float*)alloc((size_t)32 * RGBCIN[r2] * 4);
  float* rsig[9];
  for (int j = 0; j < 9; ++j) rsig[j] = (float*)alloc((size_t)32 * COUT[j] * 4);
  float* wsq = (float*)alloc((size_t)512 * 512 * 4);       // reused per block (stream-serial)
  unsigned short* wst[9];
  for (int j = 0; j < 9; ++j) wst[j] = (unsigned short*)alloc((size_t)9 * CIN[j] * COUT[j] * 2);
  unsigned short* xq = (unsigned short*)alloc((size_t)32 * 64 * 64 * 64 * 2);
  float* xbuf[2];
  xbuf[0] = (float*)alloc((size_t)32 * 64 * 64 * 64 * 4);
  xbuf[1] = (float*)alloc((size_t)32 * 64 * 64 * 64 * 4);
  float* rgbb[2];
  rgbb[0] = (float*)alloc((size_t)32 * 32 * 32 * 3 * 4);
  rgbb[1] = (float*)alloc((size_t)32 * 32 * 32 * 3 * 4);

  const int T = 256;
  // ---- styles ----
  for (int j = 0; j < 9; ++j) {
    int nthr = 32 * CIN[j];
    style_kernel<<<(nthr + T - 1) / T, T, 0, stream>>>(lat, bStyW[j], bStyB[j], sBlk[j], CIN[j]);
  }
  for (int r2 = 0; r2 < 5; ++r2) {
    int nthr = 32 * RGBCIN[r2];
    style_kernel<<<(nthr + T - 1) / T, T, 0, stream>>>(lat, rStyW[r2], rStyB[r2], sRgb[r2], RGBCIN[r2]);
  }
  // ---- demod 1/sigma + staged bf16 weights ----
  for (int j = 0; j < 9; ++j) {
    float wsc = 1.0f / sqrtf((float)(CIN[j] * 9));
    int nio = CIN[j] * COUT[j];
    wsumsq_kernel<<<(nio + T - 1) / T, T, 0, stream>>>(bConv[j], wsq, CIN[j], COUT[j], wsc * wsc);
    int nbo = 32 * COUT[j];
    rsigma_kernel<<<(nbo + T - 1) / T, T, 0, stream>>>(sBlk[j], wsq, rsig[j], CIN[j], COUT[j]);
    wstage_kernel<<<(9 * nio + T - 1) / T, T, 0, stream>>>(bConv[j], wst[j], CIN[j], COUT[j], wsc);
  }
  // ---- synthesis chain ----
  const float* xin = nullptr;
  for (int j = 0; j < 9; ++j) {
    int H = HH[j], W = H, Ci = CIN[j], Co = COUT[j];
    int mode = (j == 0) ? 0 : (UP[j] ? 2 : 1);
    const float* src = (j == 0) ? cst : xin;
    int ntot = 32 * H * W * Ci;
    xstage_kernel<<<(ntot + T - 1) / T, T, 0, stream>>>(src, sBlk[j], xq, H, W, Ci, mode);
    float* xout = xbuf[j & 1];
    int tiles = 32 * (H * W / 16) * (Co / 64);   // 16M x 64N per wave
    conv_wmma_kernel<<<(tiles + 3) / 4, 128, 0, stream>>>(
        xq, wst[j], rsig[j], bNstr[j], bBias[j], noiseP[j], xout, H, W, Ci, Co);
    xin = xout;
    if ((j & 1) == 0) {
      int r2 = j >> 1;
      float* dst = (r2 == 4) ? (float*)d_out : rgbb[r2 & 1];
      const float* prev = (r2 > 0) ? rgbb[(r2 - 1) & 1] : nullptr;
      float rwsc = 1.0f / sqrtf((float)RGBCIN[r2]);
      int nrgb = 32 * H * W * 3;
      torgb_kernel<<<(nrgb + T - 1) / T, T, 0, stream>>>(
          xin, sRgb[r2], rConv[r2], rBias[r2], prev, dst, H, W, RGBCIN[r2], rwsc, r2 > 0);
    }
  }
}
